// TensorProductConvLayer_9191230013567
// MI455X (gfx1250) — matrix operand order repaired and checked by reading, not compiled, not run on hardware
//
#include <hip/hip_runtime.h>
#include <stdint.h>

// ---------------- problem constants ----------------
#define NN      10000
#define EE      50000
#define NS      48
#define NV      10
#define NEF     128
#define HID     128
#define W1N     (NS*NS)           // 2304
#define W2N     (NS*NV)           // 480
#define W3N     (NV*NV)           // 100
#define W4N     (NV*NS)           // 480
#define WTOT    (W1N+W2N+W3N+W4N) // 3364
#define OUTC    (NS + 3*NV)       // 78
#define NT2     211               // ceil(3364/16)
#define NT1     8                 // 128/16
#define TM      16                // edges per block
#define MULW    108               // per-edge multiplier table width (107 used)
#define EPSF    1e-5f
#define INV_SQRT3 0.5773502691896258f
#define NORM01  0.13130643285972254f   // 1/sqrt(58)

#define AS1 __attribute__((address_space(1)))
#define AS3 __attribute__((address_space(3)))

typedef int v4i __attribute__((vector_size(16)));   // matches builtin param type

#if __has_builtin(__builtin_amdgcn_global_load_async_to_lds_b128) && \
    __has_builtin(__builtin_amdgcn_s_wait_asynccnt)
#define HAVE_ASYNC_LDS 1
#else
#define HAVE_ASYNC_LDS 0
#endif

typedef __attribute__((ext_vector_type(16))) __bf16 v16bf;
typedef __attribute__((ext_vector_type(2)))  __bf16 v2bf;
typedef __attribute__((ext_vector_type(8)))  float  v8f;

union BFrag {
  v16bf    bf;
  uint32_t u[8];
  uint4    q[2];
};

__device__ __forceinline__ unsigned short f2bf(float f) {
  unsigned int u = __float_as_uint(f);
  unsigned int r = (u + 0x7fffu + ((u >> 16) & 1u)) >> 16;
  return (unsigned short)r;
}

__device__ __forceinline__ uint32_t pack2bf(float a, float b) {
#if __has_builtin(__builtin_amdgcn_cvt_pk_bf16_f32)
  union { v2bf v; uint32_t u; } x;
  x.v = __builtin_amdgcn_cvt_pk_bf16_f32(a, b);
  return x.u;
#else
  return (uint32_t)f2bf(a) | ((uint32_t)f2bf(b) << 16);
#endif
}

__device__ __forceinline__ void atomAddShared(float* p, float v) {
  __hip_atomic_fetch_add(p, v, __ATOMIC_RELAXED, __HIP_MEMORY_SCOPE_WORKGROUP);
}
__device__ __forceinline__ void atomAddGlobal(float* p, float v) {
  __hip_atomic_fetch_add(p, v, __ATOMIC_RELAXED, __HIP_MEMORY_SCOPE_AGENT);
}

// ---------------------------------------------------------------------------
// Pack a row-major f32 weight matrix W[128][ncols] into bf16 WMMA B-fragment
// layout: out[((tile*4 + kstep)*32 + lane)*8 + j] packs (K=2j, 2j+1) for
// column n = tile*16 + (lane&15), K-half = (lane>>4)*16, kstep covers K=32.
// ---------------------------------------------------------------------------
__global__ void pack_wmma_b(const float* __restrict__ W, uint32_t* __restrict__ out,
                            int ncols, int total) {
  int id = blockIdx.x * blockDim.x + threadIdx.x;
  if (id >= total) return;
  int j    = id & 7;
  int lane = (id >> 3) & 31;
  int ks   = (id >> 8) & 3;
  int tile = id >> 10;
  int n = tile * 16 + (lane & 15);
  int k = ks * 32 + ((lane >> 4) * 16) + 2 * j;
  float v0 = 0.f, v1 = 0.f;
  if (n < ncols) {
    v0 = W[(size_t)k * ncols + n];
    v1 = W[(size_t)(k + 1) * ncols + n];
  }
  out[id] = pack2bf(v0, v1);
}

// ---------------------------------------------------------------------------
// Per-column consume descriptor:
//   bits [6:0]  srcOff into mulE (w1: u, w2: 48+u, w4: 96+u, w3: u)
//   bits [13:7] target index into accT
//   bit  [14]   w3 flag (3-component vector update)
//   -1          padding column (skip)
// ---------------------------------------------------------------------------
__global__ void build_desc(int* __restrict__ desc, int total) {
  int col = blockIdx.x * blockDim.x + threadIdx.x;
  if (col >= total) return;
  int d;
  if (col >= WTOT) {
    d = -1;
  } else if (col < W1N) {                      // w1[u][v] -> out0[v]
    int u = col / NS, v = col - u * NS;
    d = u | (v << 7);
  } else if (col < W1N + W2N) {                // w2[u][v] -> t2[v]
    int cc = col - W1N;
    int u = cc / NV, v = cc - u * NV;
    d = (48 + u) | ((NS + v) << 7);
  } else if (col < W1N + W2N + W3N) {          // w3[u][v] -> out1[v][0..2]
    int cc = col - (W1N + W2N);
    int u = cc / NV, v = cc - u * NV;
    d = u | ((58 + v * 3) << 7) | (1 << 14);
  } else {                                     // w4[u][v] -> out0[v] (via dot)
    int cc = col - (W1N + W2N + W3N);
    int u = cc / NS, v = cc - u * NS;
    d = (96 + u) | (v << 7);
  }
  desc[col] = d;
}

// ---------------------------------------------------------------------------
// Main fused kernel: 16 edges per block, 256 threads (8 wave32).
//   edge_attr tile staged to LDS via GLOBAL_LOAD_ASYNC_TO_LDS_B128 (ASYNCcnt)
//   GEMM1 (WMMA bf16): h = relu(edge_attr @ fc_w1 + b1)          [16 x 128]
//   GEMM2 (WMMA bf16): w-tile = h @ fc_w2 + b2, consumed on the fly into
//   per-edge tensor-product accumulators in LDS, then scattered with
//   global float atomics into summed[edge_src].
// ---------------------------------------------------------------------------
__global__ __launch_bounds__(256) void tpconv_kernel(
    const float* __restrict__ node_attr, const int* __restrict__ edge_index,
    const float* __restrict__ edge_attr, const float* __restrict__ edge_sh,
    const float* __restrict__ fc_b1, const float* __restrict__ fc_b2,
    const uint32_t* __restrict__ pk1, const uint32_t* __restrict__ pk2,
    const int* __restrict__ descG,
    float* __restrict__ summed, float* __restrict__ cnt) {

  __shared__ __align__(16) float    eaF[TM * NEF];  // raw f32 edge_attr (async DMA)
  __shared__ __align__(16) uint32_t aA[TM * 64];    // edge_attr bf16 x2 per dword
  __shared__ __align__(16) uint32_t hB[TM * 64];    // hidden activations, bf16 x2
  __shared__ float sIn[TM][NS];
  __shared__ float vIn[TM][3 * NV];
  __shared__ float sh0s[TM];
  __shared__ float sh1s[TM][3];
  __shared__ float mulE[TM][MULW];                  // fused per-edge multipliers
  __shared__ float accT[TM][88];                    // [0,48)=out0 [48,58)=t2 [58,88)=out1
  __shared__ int   srcS[TM];

  const int tid  = threadIdx.x;
  const int lane = tid & 31;
  const int wid  = tid >> 5;
  const int half = lane >> 4;
  const int nIn  = lane & 15;
  const int eb   = blockIdx.x * TM;

  // ---- stage edge_attr tile into LDS (async DMA on CDNA5) ----
#if HAVE_ASYNC_LDS
  for (int t = tid; t < TM * NEF / 4; t += 256) {
    __builtin_amdgcn_global_load_async_to_lds_b128(
        (AS1 v4i*)(edge_attr + (size_t)eb * NEF + t * 4),
        (AS3 v4i*)(eaF + t * 4), 0, 0);
  }
#else
  for (int t = tid; t < TM * NEF; t += 256) {
    eaF[t] = edge_attr[(size_t)eb * NEF + t];
  }
#endif

  // prefetch next block's edge_attr tile (global_prefetch_b8)
  if (blockIdx.x + 1 < gridDim.x) {
    __builtin_prefetch(edge_attr + (size_t)(eb + TM) * NEF + tid * 8, 0, 0);
  }

  // ---- gather per-edge node features while the DMA runs ----
  for (int t = tid; t < TM * OUTC; t += 256) {
    int e = t / OUTC, j = t - e * OUTC;
    int dst = edge_index[EE + eb + e];
    float val = node_attr[(size_t)dst * OUTC + j];
    if (j < NS) sIn[e][j] = val; else vIn[e][j - NS] = val;
  }
  if (tid < TM * 4) {
    int e = tid >> 2, j = tid & 3;
    float v = edge_sh[(size_t)(eb + e) * 4 + j];
    if (j == 0) sh0s[e] = v; else sh1s[e][j - 1] = v;
  }
  if (tid < TM) srcS[tid] = edge_index[eb + tid];
  for (int t = tid; t < TM * 88; t += 256) accT[t / 88][t % 88] = 0.f;

#if HAVE_ASYNC_LDS
  __builtin_amdgcn_s_wait_asynccnt(0);
#endif
  __syncthreads();

  // ---- one-pass bf16 pack of the edge_attr tile (4 dwords/thread) ----
  for (int t = tid; t < TM * 64; t += 256) {
    int m = t >> 6, dw = t & 63;
    aA[t] = pack2bf(eaF[m * NEF + 2 * dw], eaF[m * NEF + 2 * dw + 1]);
  }

  // ---- fused per-edge multipliers:
  //  [0,48)  = sIn[u]*sh0*NORM     (w1)
  //  [48,96) = sIn[u]*NORM         (w2)
  //  [96,106)= dot[u]*NORM         (w4, dot folded inline)
  //  [106]   = sh0*NORM            (w3 row scale)
  for (int t = tid; t < TM * 107; t += 256) {
    int m = t / 107, u = t - m * 107;
    float val;
    if (u < 48) {
      val = sIn[m][u] * sh0s[m] * NORM01;
    } else if (u < 96) {
      val = sIn[m][u - 48] * NORM01;
    } else if (u < 106) {
      int uu = u - 96;
      val = NORM01 * INV_SQRT3 * (vIn[m][uu * 3 + 0] * sh1s[m][0] +
                                  vIn[m][uu * 3 + 1] * sh1s[m][1] +
                                  vIn[m][uu * 3 + 2] * sh1s[m][2]);
    } else {
      val = sh0s[m] * NORM01;
    }
    mulE[m][u] = val;
  }
  __syncthreads();

  // ---- GEMM1: each wave computes one 16x16 n-tile of h ----
  {
    BFrag a1[4];
    for (int ks = 0; ks < 4; ks++) {
      int bd = ks * 16 + half * 8;
      a1[ks].q[0] = *(const uint4*)&aA[nIn * 64 + bd];
      a1[ks].q[1] = *(const uint4*)&aA[nIn * 64 + bd + 4];
    }
    int col = wid * 16 + nIn;
    float bias = fc_b1[col];
    v8f c;
    for (int i = 0; i < 8; i++) c[i] = bias;
    for (int ks = 0; ks < 4; ks++) {
      BFrag bf;
      const uint4* bp = (const uint4*)(pk1 + (((wid * 4 + ks) * 32 + lane) << 3));
      bf.q[0] = bp[0];
      bf.q[1] = bp[1];
      c = __builtin_amdgcn_wmma_f32_16x16x32_bf16(false, a1[ks].bf, false, bf.bf,
                                                  (short)0, c, false, false);
    }
    unsigned short* h16 = (unsigned short*)hB;
    for (int r = 0; r < 8; r++) {
      float f = fmaxf(c[r], 0.f);                 // relu
      h16[(r + 8 * half) * HID + col] = f2bf(f);
    }
  }
  __syncthreads();

  // ---- GEMM2 + fused tensor-product consume ----
  {
    BFrag a2[4];
    for (int ks = 0; ks < 4; ks++) {
      int bd = ks * 16 + half * 8;
      a2[ks].q[0] = *(const uint4*)&hB[nIn * 64 + bd];
      a2[ks].q[1] = *(const uint4*)&hB[nIn * 64 + bd + 4];
    }
    const int e0 = 8 * half;
    for (int jn = wid; jn < NT2; jn += 8) {
      int col = jn * 16 + nIn;
      float bias = (col < WTOT) ? fc_b2[col] : 0.f;
      int d = descG[col];
      v8f c;
      for (int i = 0; i < 8; i++) c[i] = bias;
      for (int ks = 0; ks < 4; ks++) {
        BFrag bf;
        const uint4* bp = (const uint4*)(pk2 + (((jn * 4 + ks) * 32 + lane) << 3));
        bf.q[0] = bp[0];
        bf.q[1] = bp[1];
        c = __builtin_amdgcn_wmma_f32_16x16x32_bf16(false, a2[ks].bf, false, bf.bf,
                                                    (short)0, c, false, false);
      }
      if (d >= 0) {
        int srcOff = d & 127;
        int tgt    = (d >> 7) & 127;
        if (d & (1 << 14)) {                       // w3: 3-component update
          int ub = srcOff * 3;
          for (int r = 0; r < 8; r++) {
            int m = e0 + r;
            float wv = c[r] * mulE[m][106];
            atomAddShared(&accT[m][tgt + 0], wv * vIn[m][ub + 0]);
            atomAddShared(&accT[m][tgt + 1], wv * vIn[m][ub + 1]);
            atomAddShared(&accT[m][tgt + 2], wv * vIn[m][ub + 2]);
          }
        } else {                                   // w1/w2/w4: single FMA
          for (int r = 0; r < 8; r++) {
            int m = e0 + r;
            atomAddShared(&accT[m][tgt], c[r] * mulE[m][srcOff]);
          }
        }
      }
    }
  }
  __syncthreads();

  // ---- scatter tp into summed[src] ----
  for (int t = tid; t < TM * OUTC; t += 256) {
    int e = t / OUTC, j = t - e * OUTC;
    float val;
    if (j < NS) {
      val = accT[e][j];
    } else {
      int jj = j - NS;
      int v = jj / 3, i = jj - v * 3;
      val = accT[e][58 + jj] + accT[e][NS + v] * sh1s[e][i];
    }
    atomAddGlobal(&summed[(size_t)srcS[e] * OUTC + j], val);
  }
  if (tid < TM) atomAddGlobal(&cnt[srcS[tid]], 1.0f);
}

// ---------------------------------------------------------------------------
// Node finalize: pre = summed/max(cnt,1) + node_attr (stored back into summed)
// plus block-reduced batch-norm statistics into stats[0..105].
// ---------------------------------------------------------------------------
__global__ __launch_bounds__(256) void finalize_nodes(
    float* __restrict__ pre, const float* __restrict__ cnt,
    const float* __restrict__ node_attr, float* __restrict__ stats, int n_nodes) {
  __shared__ float s_sum[NS], s_sq[NS], s_v[NV];
  int tid = threadIdx.x;
  if (tid < NS) { s_sum[tid] = 0.f; s_sq[tid] = 0.f; }
  if (tid < NV) s_v[tid] = 0.f;
  __syncthreads();
  int n = blockIdx.x * blockDim.x + tid;
  if (n < n_nodes) {
    float c = cnt[n];
    float inv = 1.0f / fmaxf(c, 1.0f);
    for (int j = 0; j < OUTC; j++) {
      size_t idx = (size_t)n * OUTC + j;
      float val = pre[idx] * inv + node_attr[idx];
      pre[idx] = val;
      if (j < NS) {
        atomAddShared(&s_sum[j], val);
        atomAddShared(&s_sq[j], val * val);
      } else {
        atomAddShared(&s_v[(j - NS) / 3], val * val);
      }
    }
  }
  __syncthreads();
  if (tid < NS) {
    atomAddGlobal(&stats[tid], s_sum[tid]);
    atomAddGlobal(&stats[NS + tid], s_sq[tid]);
  }
  if (tid < NV) atomAddGlobal(&stats[96 + tid], s_v[tid]);
}

// stats[106..153]=mean_s, [154..201]=scale_s, [202..211]=scale_v
__global__ void bn_reduce(float* __restrict__ stats, const float* __restrict__ bn_w,
                          int n_nodes) {
  int j = threadIdx.x;
  float invN = 1.0f / (float)n_nodes;
  if (j < NS) {
    float mean = stats[j] * invN;
    float var = stats[NS + j] * invN - mean * mean;
    stats[106 + j] = mean;
    stats[154 + j] = rsqrtf(var + EPSF) * bn_w[j];
  } else if (j < NS + NV) {
    int u = j - NS;
    float vn = stats[96 + u] * invN * (1.0f / 3.0f);
    stats[202 + u] = rsqrtf(vn + EPSF) * bn_w[NS + u];
  }
}

__global__ void bn_apply(const float* __restrict__ pre, const float* __restrict__ stats,
                         const float* __restrict__ bn_b, float* __restrict__ out,
                         int total) {
  int idx = blockIdx.x * blockDim.x + threadIdx.x;
  if (idx >= total) return;
  int j = idx % OUTC;
  float v = pre[idx];
  if (j < NS) out[idx] = (v - stats[106 + j]) * stats[154 + j] + bn_b[j];
  else        out[idx] = v * stats[202 + (j - NS) / 3];
}

// ---------------------------------------------------------------------------
extern "C" void kernel_launch(void* const* d_in, const int* in_sizes, int n_in,
                              void* d_out, int out_size, void* d_ws, size_t ws_size,
                              hipStream_t stream) {
  (void)in_sizes; (void)n_in; (void)out_size; (void)ws_size;
  const float* node_attr  = (const float*)d_in[0];
  const int*   edge_index = (const int*)d_in[1];
  const float* edge_attr  = (const float*)d_in[2];
  const float* edge_sh    = (const float*)d_in[3];
  const float* fc_w1      = (const float*)d_in[4];
  const float* fc_b1      = (const float*)d_in[5];
  const float* fc_w2      = (const float*)d_in[6];
  const float* fc_b2      = (const float*)d_in[7];
  const float* bn_w       = (const float*)d_in[8];
  const float* bn_b       = (const float*)d_in[9];
  float* out = (float*)d_out;

  // workspace layout
  char* ws = (char*)d_ws;
  const size_t w1p_bytes  = (size_t)NT1 * 4 * 32 * 8 * 4;   //  32768
  const size_t w2p_bytes  = (size_t)NT2 * 4 * 32 * 8 * 4;   // 864256
  const size_t desc_bytes = (size_t)NT2 * 16 * 4;           //  13504
  uint32_t* pk1   = (uint32_t*)ws;
  uint32_t* pk2   = (uint32_t*)(ws + w1p_bytes);
  int*      descG = (int*)(ws + w1p_bytes + w2p_bytes);
  float* summed   = (float*)(ws + w1p_bytes + w2p_bytes + desc_bytes); // [N][78]
  float* cnt      = summed + (size_t)NN * OUTC;             // [N]
  float* stats    = cnt + NN;                               // 212 floats used

  // zero accumulators (summed + cnt + stats) in one async memset
  (void)hipMemsetAsync(summed, 0, ((size_t)NN * OUTC + NN + 256) * sizeof(float),
                       stream);

  // pack weights to bf16 WMMA-B fragment layout + build consume descriptors
  {
    int tot1 = NT1 * 4 * 32 * 8;
    int tot2 = NT2 * 4 * 32 * 8;
    int totd = NT2 * 16;
    pack_wmma_b<<<(tot1 + 255) / 256, 256, 0, stream>>>(fc_w1, pk1, HID, tot1);
    pack_wmma_b<<<(tot2 + 255) / 256, 256, 0, stream>>>(fc_w2, pk2, WTOT, tot2);
    build_desc<<<(totd + 255) / 256, 256, 0, stream>>>(descG, totd);
  }

  // fused edge pipeline
  tpconv_kernel<<<EE / TM, 256, 0, stream>>>(node_attr, edge_index, edge_attr,
                                             edge_sh, fc_b1, fc_b2, pk1, pk2,
                                             descG, summed, cnt);

  // segment-mean + residual + BN
  finalize_nodes<<<(NN + 255) / 256, 256, 0, stream>>>(summed, cnt, node_attr,
                                                       stats, NN);
  bn_reduce<<<1, 64, 0, stream>>>(stats, bn_w, NN);
  bn_apply<<<((NN * OUTC) + 255) / 256, 256, 0, stream>>>(summed, stats, bn_b,
                                                          out, NN * OUTC);
}
